// STICKYLlamaAttention_22634477650088
// MI455X (gfx1250) — matrix operand order repaired and checked by reading, not compiled, not run on hardware
//
#include <hip/hip_runtime.h>
#include <hip/hip_bf16.h>
#include <math.h>

// ---------------------------------------------------------------------------
// Types matching CDNA5 WMMA fragment shapes (wave32):
//   v_wmma_f32_16x16x32_bf16:  A = 16x32 bf16 (8 VGPR), B = 32x16 bf16 (8 VGPR),
//                              C/D = 16x16 f32 (8 VGPR)
// ---------------------------------------------------------------------------
typedef __attribute__((ext_vector_type(16))) __bf16 bf16x16;
typedef __attribute__((ext_vector_type(8)))  __bf16 bf16x8;
typedef __attribute__((ext_vector_type(8)))  float  f32x8;
typedef __attribute__((ext_vector_type(4)))  unsigned int u32x4;
typedef __attribute__((ext_vector_type(4)))  int i32x4;
typedef __attribute__((ext_vector_type(8)))  int i32x8;

#define S_LEN   2048
#define HDIM    4096
#define N_HEADS 32
#define N_KV    8
#define DH      128
#define NTOT_QKV 6144   // 4096 (Q) + 1024 (K) + 1024 (V)

// ---------------------------------------------------------------------------
// Tensor Data Mover support (guarded; falls back to manual staging).
// ---------------------------------------------------------------------------
#ifdef __has_builtin
#if __has_builtin(__builtin_amdgcn_tensor_load_to_lds) && \
    __has_builtin(__builtin_amdgcn_s_wait_tensorcnt)
#define USE_TDM 1
#endif
#endif

#ifdef USE_TDM
// 2-D tile DMA: global (row-major, 2-byte elems, row stride 'stride0' elems)
// -> LDS at byte offset lds_off, rows stored back-to-back (tile_d0 elems/row).
// D# layout per CDNA5 ISA (count=1, type=2=image).
static __device__ __forceinline__ void tdm_load_2d_bf16(
    unsigned int lds_off, const void* gaddr, unsigned int tensor_d0,
    unsigned int tensor_d1, unsigned int stride0, unsigned int tile_d0,
    unsigned int tile_d1) {
  unsigned long long ga = (unsigned long long)gaddr;
  u32x4 g0 = {1u,                                  // [1:0] count=1, rest 0
              lds_off,                             // [63:32] lds_addr (bytes)
              (unsigned int)ga,                    // global_addr[31:0]
              (unsigned int)((ga >> 32) & 0x1FFFFFFull) | (2u << 30)}; // +type=2
  i32x8 g1 = {
      (int)(1u << 16),                                      // data_size=2B
      (int)((tensor_d0 & 0xFFFFu) << 16),                   // td0[15:0]
      (int)((tensor_d0 >> 16) | ((tensor_d1 & 0xFFFFu) << 16)),
      (int)((tensor_d1 >> 16) | (tile_d0 << 16)),           // tile_dim0
      (int)(tile_d1 & 0xFFFFu),                             // tile_dim1
      (int)stride0,                                         // dim0_stride lo32
      0, 0};
  i32x4 g2 = {0, 0, 0, 0};
  i32x4 g3 = {0, 0, 0, 0};
#if __clang_major__ >= 23
  i32x8 g4 = {0, 0, 0, 0, 0, 0, 0, 0};
  __builtin_amdgcn_tensor_load_to_lds(g0, g1, g2, g3, g4, 0);
#else
  __builtin_amdgcn_tensor_load_to_lds(g0, g1, g2, g3, 0);
#endif
}
#endif

static __device__ __forceinline__ f32x8 wmma_bf16(bf16x16 a, bf16x16 b, f32x8 c) {
  return __builtin_amdgcn_wmma_f32_16x16x32_bf16(false, a, false, b, (short)0, c,
                                                 false, false);
}

// ---------------------------------------------------------------------------
__global__ void cvt_f32_bf16_kernel(const float* __restrict__ x,
                                    __bf16* __restrict__ y, int n) {
  int i = blockIdx.x * 256 + threadIdx.x;
  if (i < n) y[i] = (__bf16)x[i];
}

__global__ void zero_f32_kernel(float* __restrict__ p, int n) {
  int i = blockIdx.x * 256 + threadIdx.x;
  if (i < n) p[i] = 0.0f;
}

// ---------------------------------------------------------------------------
// GEMM: Y[m][n] = sum_k A[m][k] * W[n][k]  (A bf16 row-major, W fp32 row-major)
// 128x128 block tile, K-step 32, double-buffered LDS software pipeline.
// 256 threads = 8 waves (4x2), each wave: 32x64 sub-tile = 8 WMMAs / K-step.
//  mode 0: fp32 -> outF                    (o_proj -> d_out)
//  mode 1: bf16 scatter -> Qb[h][s][d], Kt[kv][d][s] (transposed!), Vb[kv][s][d]
// ---------------------------------------------------------------------------
__global__ __launch_bounds__(256) void gemm_kernel(
    const __bf16* __restrict__ A, const float* __restrict__ W0,
    const float* __restrict__ W1, const float* __restrict__ W2, int mode,
    float* __restrict__ outF, __bf16* __restrict__ Qb, __bf16* __restrict__ Kt,
    __bf16* __restrict__ Vb) {
  __shared__ __attribute__((aligned(32))) __bf16 lds_a[2][128 * 32];  // [m][k]
  __shared__ __attribute__((aligned(32))) __bf16 lds_b[2][32 * 128];  // [k][n]

  const int tid  = threadIdx.x;
  const int lane = tid & 31, wave = tid >> 5;
  const int wr = wave >> 1, wc = wave & 1;
  const int half = lane >> 4, lr = lane & 15;
  const int m0 = blockIdx.y * 128;
  const int n0 = blockIdx.x * 128;

  f32x8 acc[2][4];
#pragma unroll
  for (int mi = 0; mi < 2; mi++)
#pragma unroll
    for (int ni = 0; ni < 4; ni++)
#pragma unroll
      for (int e = 0; e < 8; e++) acc[mi][ni][e] = 0.0f;

  // ---- staging assignments ----
  // A: one bf16x16 (32 B) per thread
  const int arow = tid >> 1, akp = (tid & 1) << 4;
  // W: n-pair x 8 k per thread -> 8 packed ds_store_b32
  const int np = (tid & 63) << 1;     // 0..126 even
  const int kh = tid >> 6;            // 0..3 -> k = kh*8 .. +7
  const int ng = n0 + np;
  const float* Wp;
  int wrow;
  if (ng < 4096)      { Wp = W0; wrow = ng; }
  else if (ng < 5120) { Wp = W1; wrow = ng - 4096; }
  else                { Wp = W2; wrow = ng - 5120; }
  const float* wr0base = Wp + (size_t)wrow * HDIM + kh * 8;

  const int KT = HDIM / 32;           // 128 K-steps
  bf16x16 aReg;
  float w0r[8], w1r[8];

  // ---- prologue: stage tile 0 ----
#ifdef USE_TDM
  if (wave == 0)
    tdm_load_2d_bf16((unsigned int)(unsigned long long)(const void*)&lds_a[0][0],
                     A + (size_t)m0 * HDIM, HDIM, S_LEN, HDIM, 32, 128);
#else
  aReg = *(const bf16x16*)(A + (size_t)(m0 + arow) * HDIM + akp);
  *(bf16x16*)&lds_a[0][arow * 32 + akp] = aReg;
#endif
  {
    const float* r0 = wr0base;
#pragma unroll
    for (int i = 0; i < 8; i++) { w0r[i] = r0[i]; w1r[i] = r0[HDIM + i]; }
#pragma unroll
    for (int i = 0; i < 8; i++) {
      union { __bf16 h[2]; unsigned int u; } p;
      p.h[0] = (__bf16)w0r[i]; p.h[1] = (__bf16)w1r[i];
      *(unsigned int*)&lds_b[0][(kh * 8 + i) * 128 + np] = p.u;
    }
  }

  for (int t = 0; t < KT; ++t) {
#ifdef USE_TDM
    if (wave == 0) __builtin_amdgcn_s_wait_tensorcnt(0);
#endif
    __syncthreads();
    const int nxt = t + 1;
    const int k0n = nxt * 32;
    if (nxt < KT) {
#ifdef USE_TDM
      if (wave == 0)
        tdm_load_2d_bf16(
            (unsigned int)(unsigned long long)(const void*)&lds_a[nxt & 1][0],
            A + (size_t)m0 * HDIM + k0n, HDIM, S_LEN, HDIM, 32, 128);
#else
      aReg = *(const bf16x16*)(A + (size_t)(m0 + arow) * HDIM + k0n + akp);
#endif
      const float* r0 = wr0base + k0n;
      __builtin_prefetch(r0 + 32, 0, 0);
#pragma unroll
      for (int i = 0; i < 8; i++) { w0r[i] = r0[i]; w1r[i] = r0[HDIM + i]; }
    }

    // ---- compute from buffer t&1 ----
    {
      const __bf16* la = &lds_a[t & 1][0];
      const __bf16* lb = &lds_b[t & 1][0];
      bf16x16 af[2];
#pragma unroll
      for (int mi = 0; mi < 2; mi++) {
        const int row = wr * 32 + mi * 16 + lr;
        bf16x8 lo = *(const bf16x8*)&la[row * 32 + half * 8];
        bf16x8 hi = *(const bf16x8*)&la[row * 32 + 16 + half * 8];
#pragma unroll
        for (int e = 0; e < 8; e++) { af[mi][e] = lo[e]; af[mi][e + 8] = hi[e]; }
      }
#pragma unroll
      for (int ni = 0; ni < 4; ni++) {
        bf16x16 bf;
#pragma unroll
        for (int e = 0; e < 8; e++) {
          bf[e]     = lb[(half * 8 + e) * 128 + wc * 64 + ni * 16 + lr];
          bf[e + 8] = lb[(16 + half * 8 + e) * 128 + wc * 64 + ni * 16 + lr];
        }
#pragma unroll
        for (int mi = 0; mi < 2; mi++)
          acc[mi][ni] = wmma_bf16(af[mi], bf, acc[mi][ni]);
      }
    }

    if (nxt < KT) {
#ifndef USE_TDM
      *(bf16x16*)&lds_a[nxt & 1][arow * 32 + akp] = aReg;
#endif
#pragma unroll
      for (int i = 0; i < 8; i++) {
        union { __bf16 h[2]; unsigned int u; } p;
        p.h[0] = (__bf16)w0r[i]; p.h[1] = (__bf16)w1r[i];
        *(unsigned int*)&lds_b[nxt & 1][(kh * 8 + i) * 128 + np] = p.u;
      }
    }
  }

  // ---- epilogue ----
#pragma unroll
  for (int mi = 0; mi < 2; mi++) {
#pragma unroll
    for (int ni = 0; ni < 4; ni++) {
      const int n = n0 + wc * 64 + ni * 16 + lr;
#pragma unroll
      for (int r = 0; r < 8; r++) {
        const int m = m0 + wr * 32 + mi * 16 + r + half * 8;
        const float v = acc[mi][ni][r];
        if (mode == 0) {
          outF[(size_t)m * HDIM + n] = v;
        } else {
          if (n < 4096) {
            const int h = n >> 7, d = n & 127;
            Qb[((size_t)h * S_LEN + m) * DH + d] = (__bf16)v;
          } else if (n < 5120) {
            const int h = (n - 4096) >> 7, d = n & 127;
            Kt[((size_t)h * DH + d) * S_LEN + m] = (__bf16)v;   // transposed
          } else {
            const int h = (n - 5120) >> 7, d = n & 127;
            Vb[((size_t)h * S_LEN + m) * DH + d] = (__bf16)v;
          }
        }
      }
    }
  }
}

// ---------------------------------------------------------------------------
// Llama3-scaled RoPE in place. Q: head-major [h][s][d]; K: transposed [kv][d][s].
// grid = (S, 40): y < 32 -> Q head, else KV head. 64 threads = rotation half.
// ---------------------------------------------------------------------------
__global__ void rope_kernel(__bf16* __restrict__ Qb, __bf16* __restrict__ Kt) {
  const int d  = threadIdx.x;   // 0..63
  const int s  = blockIdx.x;
  const int hh = blockIdx.y;
  float inv = __expf(-((float)(2 * d) / 128.0f) * 13.122363377404328f);
  float wavelen = 6.2831853071795865f / inv;
  float smooth = (8192.0f / wavelen - 1.0f) / 3.0f;
  smooth = fminf(fmaxf(smooth, 0.0f), 1.0f);
  float nif = (1.0f - smooth) * (inv * 0.125f) + smooth * inv;
  float ang = (float)s * nif;
  float c = cosf(ang), sn = sinf(ang);
  if (hh < 32) {
    __bf16* base = Qb + ((size_t)hh * S_LEN + s) * DH;
    float x1 = (float)base[d], x2 = (float)base[d + 64];
    base[d]      = (__bf16)(x1 * c - x2 * sn);
    base[d + 64] = (__bf16)(x2 * c + x1 * sn);
  } else {
    __bf16* base = Kt + (size_t)(hh - 32) * DH * S_LEN;
    float x1 = (float)base[(size_t)d * S_LEN + s];
    float x2 = (float)base[(size_t)(d + 64) * S_LEN + s];
    base[(size_t)d * S_LEN + s]        = (__bf16)(x1 * c - x2 * sn);
    base[(size_t)(d + 64) * S_LEN + s] = (__bf16)(x2 * c + x1 * sn);
  }
}

// ---------------------------------------------------------------------------
// Two-pass causal flash attention. Block = (64 q rows, head); 4 waves x 16 rows.
// K tiles (32 keys x 128 d, already transposed in global) staged into LDS by
// the Tensor Data Mover (fallback: contiguous 64 B row copies).
// ---------------------------------------------------------------------------
__global__ __launch_bounds__(128) void attn_kernel(
    const __bf16* __restrict__ Qb, const __bf16* __restrict__ Kt,
    const __bf16* __restrict__ Vb, __bf16* __restrict__ Ab,
    float* __restrict__ accOut) {
  __shared__ __attribute__((aligned(32))) __bf16 lds_kt[128 * 32];  // [d][key]
  __shared__ __attribute__((aligned(32))) __bf16 lds_p[4 * 16 * 32];
  __shared__ float acc_lds[S_LEN];

  const int tid  = threadIdx.x;
  const int lane = tid & 31, wave = tid >> 5;
  const int half = lane >> 4, lr = lane & 15;
  const int qb   = blockIdx.x;
  const int head = blockIdx.y;
  const int kv   = head >> 2;
  const int q0   = qb * 64 + wave * 16;
  const int ktMax = 2 * qb + 2;

  for (int i = tid; i < S_LEN; i += 128) acc_lds[i] = 0.0f;

  const float qscale = 0.08838834764831845f;  // 1/sqrt(128)
  bf16x16 qf[4];
  {
    const __bf16* qbase = Qb + ((size_t)head * S_LEN + q0 + lr) * DH;
#pragma unroll
    for (int ks = 0; ks < 4; ks++) {
      bf16x8 lo = *(const bf16x8*)(qbase + ks * 32 + half * 8);
      bf16x8 hi = *(const bf16x8*)(qbase + ks * 32 + 16 + half * 8);
#pragma unroll
      for (int e = 0; e < 8; e++) {
        qf[ks][e]     = (__bf16)((float)lo[e] * qscale);
        qf[ks][e + 8] = (__bf16)((float)hi[e] * qscale);
      }
    }
  }

  float mrow[8], lrow[8], inv_l[8];
#pragma unroll
  for (int r = 0; r < 8; r++) { mrow[r] = -1e30f; lrow[r] = 0.0f; }
  f32x8 o[8];

  const __bf16* ktBase = Kt + (size_t)kv * DH * S_LEN;

  for (int pass = 0; pass < 2; ++pass) {
    if (pass == 1) {
#pragma unroll
      for (int r = 0; r < 8; r++) inv_l[r] = 1.0f / lrow[r];
#pragma unroll
      for (int dt = 0; dt < 8; dt++)
#pragma unroll
        for (int e = 0; e < 8; e++) o[dt][e] = 0.0f;
    }
    for (int kt = 0; kt < ktMax; ++kt) {
      const int key0 = kt * 32;
      __syncthreads();
#ifdef USE_TDM
      if (wave == 0) {
        tdm_load_2d_bf16((unsigned int)(unsigned long long)(const void*)&lds_kt[0],
                         ktBase + key0, S_LEN, DH, S_LEN, 32, 128);
        __builtin_amdgcn_s_wait_tensorcnt(0);
      }
#else
      {  // fallback: per-thread 64 B contiguous row copy, d = tid
        const __bf16* src = ktBase + (size_t)tid * S_LEN + key0;
        *(bf16x16*)&lds_kt[tid * 32]      = *(const bf16x16*)src;
        *(bf16x16*)&lds_kt[tid * 32 + 16] = *(const bf16x16*)(src + 16);
      }
#endif
      __syncthreads();

      // scores: 16 q x 32 keys per wave
      f32x8 sfr[2];
#pragma unroll
      for (int ni = 0; ni < 2; ni++) {
        f32x8 s;
#pragma unroll
        for (int e = 0; e < 8; e++) s[e] = 0.0f;
#pragma unroll
        for (int ks = 0; ks < 4; ks++) {
          bf16x16 kf;
#pragma unroll
          for (int e = 0; e < 8; e++) {
            kf[e]     = lds_kt[(ks * 32 + half * 8 + e) * 32 + ni * 16 + lr];
            kf[e + 8] = lds_kt[(ks * 32 + 16 + half * 8 + e) * 32 + ni * 16 + lr];
          }
          s = wmma_bf16(qf[ks], kf, s);
        }
        sfr[ni] = s;
      }
      const int rowB = q0 + half * 8;
#pragma unroll
      for (int ni = 0; ni < 2; ni++) {
        const int key = key0 + ni * 16 + lr;
#pragma unroll
        for (int r = 0; r < 8; r++)
          if (key > rowB + r) sfr[ni][r] = -1e30f;
      }

      if (pass == 0) {
#pragma unroll
        for (int r = 0; r < 8; r++) {
          float v0 = sfr[0][r], v1 = sfr[1][r];
          float mx = fmaxf(v0, v1);
#pragma unroll
          for (int off = 1; off < 16; off <<= 1)
            mx = fmaxf(mx, __shfl_xor(mx, off, 32));
          float mnew = fmaxf(mrow[r], mx);
          float e = __expf(v0 - mnew) + __expf(v1 - mnew);
#pragma unroll
          for (int off = 1; off < 16; off <<= 1) e += __shfl_xor(e, off, 32);
          lrow[r] = lrow[r] * __expf(mrow[r] - mnew) + e;
          mrow[r] = mnew;
        }
      } else {
        __bf16* pw = lds_p + wave * (16 * 32);
#pragma unroll
        for (int ni = 0; ni < 2; ni++) {
          float cs = 0.0f;
#pragma unroll
          for (int r = 0; r < 8; r++) {
            float p = __expf(sfr[ni][r] - mrow[r]) * inv_l[r];
            cs += p;
            pw[(r + half * 8) * 32 + ni * 16 + lr] = (__bf16)p;
          }
          cs += __shfl_xor(cs, 16, 32);
          if (lane < 16) atomicAdd(&acc_lds[key0 + ni * 16 + lr], cs);
        }
        __syncthreads();
        bf16x16 pf;
        {
          bf16x8 lo = *(const bf16x8*)&pw[lr * 32 + half * 8];
          bf16x8 hi = *(const bf16x8*)&pw[lr * 32 + 16 + half * 8];
#pragma unroll
          for (int e = 0; e < 8; e++) { pf[e] = lo[e]; pf[e + 8] = hi[e]; }
        }
        const __bf16* vrow = Vb + ((size_t)kv * S_LEN + key0) * DH;
#pragma unroll
        for (int dt = 0; dt < 8; dt++) {
          bf16x16 vf;
#pragma unroll
          for (int e = 0; e < 8; e++) {
            vf[e]     = vrow[(half * 8 + e) * DH + dt * 16 + lr];
            vf[e + 8] = vrow[(16 + half * 8 + e) * DH + dt * 16 + lr];
          }
          o[dt] = wmma_bf16(pf, vf, o[dt]);
        }
      }
    }
  }

#pragma unroll
  for (int dt = 0; dt < 8; dt++) {
#pragma unroll
    for (int r = 0; r < 8; r++) {
      const int m = q0 + r + half * 8;
      Ab[(size_t)m * HDIM + head * DH + dt * 16 + lr] = (__bf16)o[dt][r];
    }
  }

  __syncthreads();
  const int nk = (qb + 1) * 64;
  for (int i = tid; i < nk; i += 128)
    atomicAdd(&accOut[(size_t)kv * S_LEN + i], acc_lds[i] * 0.25f);
}

// ---------------------------------------------------------------------------
// Workspace layout (56 MB):
//  [ 0,16) MB Xb : hidden bf16 [2048][4096]
//  [16,32) MB Qb : Q [32][2048][128]
//  [32,36) MB Kt : K transposed [8][128][2048]
//  [36,40) MB Vb : V [8][2048][128]
//  [40,56) MB Ab : attention out bf16 [2048][4096]
// ---------------------------------------------------------------------------
extern "C" void kernel_launch(void* const* d_in, const int* in_sizes, int n_in,
                              void* d_out, int out_size, void* d_ws,
                              size_t ws_size, hipStream_t stream) {
  const float* hs = (const float*)d_in[0];
  const float* wq = (const float*)d_in[1];
  const float* wk = (const float*)d_in[2];
  const float* wv = (const float*)d_in[3];
  const float* wo = (const float*)d_in[4];
  float* outp = (float*)d_out;
  char* ws = (char*)d_ws;

  __bf16* Xb = (__bf16*)(ws);
  __bf16* Qb = (__bf16*)(ws + (size_t)16 * 1024 * 1024);
  __bf16* Kt = (__bf16*)(ws + (size_t)32 * 1024 * 1024);
  __bf16* Vb = (__bf16*)(ws + (size_t)36 * 1024 * 1024);
  __bf16* Ab = (__bf16*)(ws + (size_t)40 * 1024 * 1024);

  const int nX = S_LEN * HDIM;
  cvt_f32_bf16_kernel<<<(nX + 255) / 256, 256, 0, stream>>>(hs, Xb, nX);

  gemm_kernel<<<dim3(NTOT_QKV / 128, S_LEN / 128), 256, 0, stream>>>(
      Xb, wq, wk, wv, /*mode=*/1, nullptr, Qb, Kt, Vb);

  rope_kernel<<<dim3(S_LEN, N_HEADS + N_KV), 64, 0, stream>>>(Qb, Kt);

  float* accOut = outp + (size_t)S_LEN * HDIM;
  zero_f32_kernel<<<(N_KV * S_LEN + 255) / 256, 256, 0, stream>>>(accOut,
                                                                  N_KV * S_LEN);

  attn_kernel<<<dim3(S_LEN / 64, N_HEADS), 128, 0, stream>>>(Qb, Kt, Vb, Ab,
                                                             accOut);

  gemm_kernel<<<dim3(HDIM / 128, S_LEN / 128), 256, 0, stream>>>(
      Ab, wo, wo, wo, /*mode=*/0, outp, nullptr, nullptr, nullptr);
}
